// QLSTM_65481071401286
// MI455X (gfx1250) — matrix operand order repaired and checked by reading, not compiled, not run on hardware
//
#include <hip/hip_runtime.h>
#include <hip/hip_bf16.h>

// ---------------------------------------------------------------------------
// Problem constants (match reference)
// ---------------------------------------------------------------------------
#define TT  256
#define BB  8
#define II  1024
#define HH  1024
#define NHH 16
#define DHH 64
#define KG  (II + HH)   // 2048
#define NG  (4 * HH)    // 4096

typedef __attribute__((ext_vector_type(16))) __bf16 v16bf;
typedef __attribute__((ext_vector_type(8)))  float  v8f;

union AFrag { unsigned int u[8]; v16bf v; };
union CFrag { float f[8]; v8f v; };

__device__ __forceinline__ unsigned short f2bf(float f) {
    unsigned int u = __float_as_uint(f);
    u += 0x7FFFu + ((u >> 16) & 1u);   // round-to-nearest-even
    return (unsigned short)(u >> 16);
}

// ---------------------------------------------------------------------------
// Elementwise helpers
// ---------------------------------------------------------------------------
__global__ void f2bf_kernel(const float* __restrict__ src,
                            unsigned short* __restrict__ dst, int n) {
    int i = blockIdx.x * 256 + threadIdx.x;
    if (i < n) dst[i] = f2bf(src[i]);
}

__global__ void zero_f32_kernel(float* p, int n) {
    int i = blockIdx.x * 256 + threadIdx.x;
    if (i < n) p[i] = 0.f;
}

// ---------------------------------------------------------------------------
// WMMA inner loop. A panel lives in LDS (bf16), weights stream from L2.
// A fragment layout per ISA 7.12.2 (16-bit A 16x32); B loaded column-major
// straight out of the row-major weight (B[k][n] = W[n][k]).
// ---------------------------------------------------------------------------
__device__ __forceinline__ void wmma_tile_accum(const unsigned short* arow,   // LDS
                                                const unsigned short* wrow,   // global
                                                int K, int laneHi, CFrag& c) {
    for (int k0 = 0; k0 < K; k0 += 32) {
        __builtin_prefetch(wrow + k0 + 128, 0, 0);   // global_prefetch_b8
        AFrag a, b;
#pragma unroll
        for (int v = 0; v < 8; ++v) {
            int kkA = ((v < 4) ? (2 * v) : (16 + 2 * (v - 4))) + (laneHi ? 8 : 0);
            a.u[v] = *(const unsigned int*)(arow + k0 + kkA);
            int kkB = ((v < 4) ? (2 * v) : (16 + 2 * (v - 4))) + (laneHi ? 8 : 0);
            b.u[v] = *(const unsigned int*)(wrow + k0 + kkB);
        }
        c.v = __builtin_amdgcn_wmma_f32_16x16x32_bf16(false, a.v, false, b.v,
                                                      (short)0, c.v, false, false);
    }
}

// ---------------------------------------------------------------------------
// Gates GEMM: z[8,4096] = [x_t | hx] (8x2048) @ W_gates^T + b_gates
// 8 waves/block share one LDS A panel (rows 0..7 real, row 8 = zeros, used
// for the padded M=8..15 fragment rows). Grid = 4096/16/8 = 32 blocks.
// ---------------------------------------------------------------------------
__global__ __launch_bounds__(256)
void wmma_gates_kernel(const float* __restrict__ x_t,
                       const float* __restrict__ hx,
                       const unsigned short* __restrict__ Wg,
                       const float* __restrict__ bias,
                       float* __restrict__ z) {
    __shared__ unsigned short Apan[9 * KG];   // 36 KB
    const int tid = threadIdx.x;
    for (int i = tid; i < 9 * KG; i += 256) {
        int r = i >> 11, cidx = i & (KG - 1);
        float v = 0.f;
        if (r < BB) v = (cidx < II) ? x_t[r * II + cidx] : hx[r * HH + (cidx - II)];
        Apan[i] = f2bf(v);
    }
    __syncthreads();

    const int wave = tid >> 5, lane = tid & 31;
    const int laneHi = lane >> 4, l16 = lane & 15;
    const int n0 = (blockIdx.x * 8 + wave) * 16;

    CFrag c;
#pragma unroll
    for (int r = 0; r < 8; ++r) c.f[r] = 0.f;

    const unsigned short* arow = Apan + (size_t)((l16 < 8) ? l16 : 8) * KG;
    const unsigned short* wrow = Wg + (size_t)(n0 + l16) * KG;
    wmma_tile_accum(arow, wrow, KG, laneHi, c);

#pragma unroll
    for (int r = 0; r < 8; ++r) {
        int m = r + 8 * laneHi;
        if (m < BB) z[(size_t)m * NG + n0 + l16] = c.f[r] + bias[n0 + l16];
    }
}

// ---------------------------------------------------------------------------
// Generic projection GEMM: out[8,1024] = A8(8x1024 f32) @ W^T + bias
// Grid = 1024/16/8 = 8 blocks x 8 waves.
// ---------------------------------------------------------------------------
__global__ __launch_bounds__(256)
void wmma_proj_kernel(const float* __restrict__ A8,
                      const unsigned short* __restrict__ W,
                      const float* __restrict__ bias,
                      float* __restrict__ out) {
    __shared__ unsigned short Apan[9 * HH];   // 18 KB
    const int tid = threadIdx.x;
    for (int i = tid; i < 9 * HH; i += 256) {
        int r = i >> 10, cidx = i & (HH - 1);
        Apan[i] = (r < BB) ? f2bf(A8[r * HH + cidx]) : (unsigned short)0;
    }
    __syncthreads();

    const int wave = tid >> 5, lane = tid & 31;
    const int laneHi = lane >> 4, l16 = lane & 15;
    const int n0 = (blockIdx.x * 8 + wave) * 16;

    CFrag c;
#pragma unroll
    for (int r = 0; r < 8; ++r) c.f[r] = 0.f;

    const unsigned short* arow = Apan + (size_t)((l16 < 8) ? l16 : 8) * HH;
    const unsigned short* wrow = W + (size_t)(n0 + l16) * HH;
    wmma_tile_accum(arow, wrow, HH, laneHi, c);

#pragma unroll
    for (int r = 0; r < 8; ++r) {
        int m = r + 8 * laneHi;
        if (m < BB) out[(size_t)m * HH + n0 + l16] = c.f[r] + bias[n0 + l16];
    }
}

// ---------------------------------------------------------------------------
// Fused K/V projection: both share the hx A panel. Grid = 16 blocks; waves
// 0..63 (global) produce Kc[t], 64..127 produce Vc[t].
// ---------------------------------------------------------------------------
__global__ __launch_bounds__(256)
void wmma_kv_kernel(const float* __restrict__ hx,
                    const unsigned short* __restrict__ Wk,
                    const unsigned short* __restrict__ Wv,
                    const float* __restrict__ bk,
                    const float* __restrict__ bv,
                    float* __restrict__ Kt, float* __restrict__ Vt) {
    __shared__ unsigned short Apan[9 * HH];
    const int tid = threadIdx.x;
    for (int i = tid; i < 9 * HH; i += 256) {
        int r = i >> 10, cidx = i & (HH - 1);
        Apan[i] = (r < BB) ? f2bf(hx[r * HH + cidx]) : (unsigned short)0;
    }
    __syncthreads();

    const int wave = tid >> 5, lane = tid & 31;
    const int laneHi = lane >> 4, l16 = lane & 15;
    const int g = blockIdx.x * 8 + wave;          // 0..127
    const bool isV = (g >= 64);
    const int n0 = (isV ? (g - 64) : g) * 16;
    const unsigned short* W = isV ? Wv : Wk;
    const float* bias = isV ? bv : bk;
    float* out = isV ? Vt : Kt;

    CFrag c;
#pragma unroll
    for (int r = 0; r < 8; ++r) c.f[r] = 0.f;

    const unsigned short* arow = Apan + (size_t)((l16 < 8) ? l16 : 8) * HH;
    const unsigned short* wrow = W + (size_t)(n0 + l16) * HH;
    wmma_tile_accum(arow, wrow, HH, laneHi, c);

#pragma unroll
    for (int r = 0; r < 8; ++r) {
        int m = r + 8 * laneHi;
        if (m < BB) out[(size_t)m * HH + n0 + l16] = c.f[r] + bias[n0 + l16];
    }
}

// ---------------------------------------------------------------------------
// Out-projection + residual: hx = h0 + (t>0 ? ctx @ Wo^T + bo : 0)
// Also writes outputs[t]. Grid = 8 blocks.
// ---------------------------------------------------------------------------
__global__ __launch_bounds__(256)
void wmma_oproj_kernel(const float* __restrict__ ctx,
                       const unsigned short* __restrict__ Wo,
                       const float* __restrict__ bo,
                       const float* __restrict__ h0,
                       float* __restrict__ hx,
                       float* __restrict__ out_t,
                       int use_attn) {
    __shared__ unsigned short Apan[9 * HH];
    const int tid = threadIdx.x;
    if (use_attn) {
        for (int i = tid; i < 9 * HH; i += 256) {
            int r = i >> 10, cidx = i & (HH - 1);
            Apan[i] = (r < BB) ? f2bf(ctx[r * HH + cidx]) : (unsigned short)0;
        }
    }
    __syncthreads();

    const int wave = tid >> 5, lane = tid & 31;
    const int laneHi = lane >> 4, l16 = lane & 15;
    const int n0 = (blockIdx.x * 8 + wave) * 16;

    CFrag c;
#pragma unroll
    for (int r = 0; r < 8; ++r) c.f[r] = 0.f;

    if (use_attn) {
        const unsigned short* arow = Apan + (size_t)((l16 < 8) ? l16 : 8) * HH;
        const unsigned short* wrow = Wo + (size_t)(n0 + l16) * HH;
        wmma_tile_accum(arow, wrow, HH, laneHi, c);
    }

#pragma unroll
    for (int r = 0; r < 8; ++r) {
        int m = r + 8 * laneHi;
        if (m < BB) {
            int n = n0 + l16;
            float val = h0[(size_t)m * HH + n];
            if (use_attn) val += c.f[r] + bo[n];
            hx[(size_t)m * HH + n]    = val;
            out_t[(size_t)m * HH + n] = val;
        }
    }
}

// ---------------------------------------------------------------------------
// LayerNorm (4 gates) + LSTM pointwise. One block per batch row.
// ---------------------------------------------------------------------------
__global__ __launch_bounds__(256)
void lstm_pointwise_kernel(const float* __restrict__ z,       // [B][4*H]
                           const float* __restrict__ ln_g,    // [4*H]
                           const float* __restrict__ ln_b,    // [4*H]
                           float* __restrict__ cx,
                           float* __restrict__ h0) {
    __shared__ float red[256];
    __shared__ float stats[8];   // mu[0..3], inv_std[4..7]
    const int b = blockIdx.x, tid = threadIdx.x;
    const float invH = 1.f / (float)HH;

    for (int k = 0; k < 4; ++k) {
        float s = 0.f, s2 = 0.f;
#pragma unroll
        for (int j = 0; j < 4; ++j) {
            float v = z[(size_t)b * NG + k * HH + tid + 256 * j];
            s += v; s2 += v * v;
        }
        red[tid] = s; __syncthreads();
        for (int o = 128; o > 0; o >>= 1) { if (tid < o) red[tid] += red[tid + o]; __syncthreads(); }
        if (tid == 0) stats[k] = red[0] * invH;
        __syncthreads();
        red[tid] = s2; __syncthreads();
        for (int o = 128; o > 0; o >>= 1) { if (tid < o) red[tid] += red[tid + o]; __syncthreads(); }
        if (tid == 0) {
            float mu  = stats[k];
            float var = red[0] * invH - mu * mu;
            stats[4 + k] = rsqrtf(var + 1e-5f);
        }
        __syncthreads();
    }

#pragma unroll
    for (int j = 0; j < 4; ++j) {
        int h = tid + 256 * j;
        float zn[4];
#pragma unroll
        for (int k = 0; k < 4; ++k) {
            float v = z[(size_t)b * NG + k * HH + h];
            zn[k] = (v - stats[k]) * stats[4 + k] * ln_g[k * HH + h] + ln_b[k * HH + h];
        }
        float f = 1.f / (1.f + __expf(-zn[0]));
        float i = 1.f / (1.f + __expf(-zn[1]));
        float g = tanhf(zn[2]);
        float o = 1.f / (1.f + __expf(-zn[3]));
        float cn = f * cx[(size_t)b * HH + h] + i * g;
        cx[(size_t)b * HH + h] = cn;
        h0[(size_t)b * HH + h] = o * tanhf(cn);
    }
}

// ---------------------------------------------------------------------------
// Attention over cached K/V for s < t. One block per (batch, head).
// ---------------------------------------------------------------------------
__global__ __launch_bounds__(64)
void attention_kernel(const float* __restrict__ q,
                      const float* __restrict__ Kc,   // [T][B][H]
                      const float* __restrict__ Vc,
                      float* __restrict__ ctx, int t) {
    __shared__ float sc[TT];
    __shared__ float red[64];
    const int b   = blockIdx.x >> 4;
    const int hh  = blockIdx.x & 15;
    const int tid = threadIdx.x;
    const float scale = 0.125f;  // 1/sqrt(64)
    const float* qv = q + (size_t)b * HH + hh * DHH;

    float lmax = -1e30f;
    for (int s = tid; s < t; s += 64) {
        const float* kv = Kc + (size_t)s * (BB * HH) + (size_t)b * HH + hh * DHH;
        float d = 0.f;
#pragma unroll
        for (int e = 0; e < DHH; ++e) d += qv[e] * kv[e];
        d *= scale;
        sc[s] = d;
        lmax = fmaxf(lmax, d);
    }
    red[tid] = lmax; __syncthreads();
    for (int o = 32; o > 0; o >>= 1) { if (tid < o) red[tid] = fmaxf(red[tid], red[tid + o]); __syncthreads(); }
    float mx = red[0];
    __syncthreads();

    float lsum = 0.f;
    for (int s = tid; s < t; s += 64) {
        float e = __expf(sc[s] - mx);
        sc[s] = e;
        lsum += e;
    }
    red[tid] = lsum; __syncthreads();
    for (int o = 32; o > 0; o >>= 1) { if (tid < o) red[tid] += red[tid + o]; __syncthreads(); }
    float inv = 1.f / red[0];
    __syncthreads();

    // tid == d within the head
    float acc = 0.f;
    for (int s = 0; s < t; ++s)
        acc += sc[s] * Vc[(size_t)s * (BB * HH) + (size_t)b * HH + hh * DHH + tid];
    ctx[(size_t)b * HH + hh * DHH + tid] = acc * inv;
}

__global__ void final_copy_kernel(const float* __restrict__ hx,
                                  const float* __restrict__ cx,
                                  float* __restrict__ dst) {
    int i = blockIdx.x * 256 + threadIdx.x;
    if (i < BB * HH) { dst[i] = hx[i]; dst[BB * HH + i] = cx[i]; }
}

// ---------------------------------------------------------------------------
// Host launcher
// ---------------------------------------------------------------------------
extern "C" void kernel_launch(void* const* d_in, const int* in_sizes, int n_in,
                              void* d_out, int out_size, void* d_ws, size_t ws_size,
                              hipStream_t stream) {
    const float* inputs  = (const float*)d_in[0];   // [T,B,I]
    const float* W_gates = (const float*)d_in[1];   // [4,H,I+H]
    const float* b_gates = (const float*)d_in[2];   // [4,H]
    const float* ln_g    = (const float*)d_in[3];
    const float* ln_b    = (const float*)d_in[4];
    const float* Wq      = (const float*)d_in[5];
    const float* Wk      = (const float*)d_in[6];
    const float* Wv      = (const float*)d_in[7];
    const float* bq      = (const float*)d_in[8];
    const float* bk      = (const float*)d_in[9];
    const float* bv      = (const float*)d_in[10];
    const float* Wo      = (const float*)d_in[11];
    const float* bo      = (const float*)d_in[12];
    float* out = (float*)d_out;

    // ---- workspace layout (bf16 weights first, then f32 buffers) ----
    unsigned short* Wg_b = (unsigned short*)d_ws;             // 4*H*KG
    unsigned short* Wq_b = Wg_b + (size_t)4 * HH * KG;
    unsigned short* Wk_b = Wq_b + (size_t)HH * HH;
    unsigned short* Wv_b = Wk_b + (size_t)HH * HH;
    unsigned short* Wo_b = Wv_b + (size_t)HH * HH;

    float* F   = (float*)(Wo_b + (size_t)HH * HH);
    float* hx  = F;                                 // B*H
    float* cx  = hx  + BB * HH;
    float* h0  = cx  + BB * HH;
    float* q   = h0  + BB * HH;
    float* ctx = q   + BB * HH;
    float* z   = ctx + BB * HH;                     // B*4H
    float* Kc  = z   + (size_t)BB * NG;             // T*B*H
    float* Vc  = Kc  + (size_t)TT * BB * HH;

    // ---- per-call weight conversion + state init ----
    {
        int n = 4 * HH * KG;
        f2bf_kernel<<<(n + 255) / 256, 256, 0, stream>>>(W_gates, Wg_b, n);
        n = HH * HH;
        f2bf_kernel<<<(n + 255) / 256, 256, 0, stream>>>(Wq, Wq_b, n);
        f2bf_kernel<<<(n + 255) / 256, 256, 0, stream>>>(Wk, Wk_b, n);
        f2bf_kernel<<<(n + 255) / 256, 256, 0, stream>>>(Wv, Wv_b, n);
        f2bf_kernel<<<(n + 255) / 256, 256, 0, stream>>>(Wo, Wo_b, n);
        zero_f32_kernel<<<(BB * HH + 255) / 256, 256, 0, stream>>>(hx, BB * HH);
        zero_f32_kernel<<<(BB * HH + 255) / 256, 256, 0, stream>>>(cx, BB * HH);
    }

    // ---- recurrent loop ----
    for (int t = 0; t < TT; ++t) {
        // z = [x_t | hx] @ W_gates^T + b_gates
        wmma_gates_kernel<<<NG / 16 / 8, 256, 0, stream>>>(
            inputs + (size_t)t * BB * II, hx, Wg_b, b_gates, z);

        lstm_pointwise_kernel<<<BB, 256, 0, stream>>>(z, ln_g, ln_b, cx, h0);

        // q = h0 @ Wq^T + bq
        wmma_proj_kernel<<<HH / 16 / 8, 256, 0, stream>>>(h0, Wq_b, bq, q);

        if (t > 0)
            attention_kernel<<<BB * NHH, 64, 0, stream>>>(q, Kc, Vc, ctx, t);

        // hx = h0 + (t>0 ? ctx @ Wo^T + bo : 0); also writes outputs[t]
        wmma_oproj_kernel<<<HH / 16 / 8, 256, 0, stream>>>(
            ctx, Wo_b, bo, h0, hx, out + (size_t)t * BB * HH, (t > 0) ? 1 : 0);

        // fused K/V cache projections of the new hidden state
        wmma_kv_kernel<<<2 * HH / 16 / 8, 256, 0, stream>>>(
            hx, Wk_b, Wv_b, bk, bv,
            Kc + (size_t)t * BB * HH, Vc + (size_t)t * BB * HH);
    }

    final_copy_kernel<<<(BB * HH + 255) / 256, 256, 0, stream>>>(
        hx, cx, out + (size_t)TT * BB * HH);
}
